// GCRN_87617332838643
// MI455X (gfx1250) — compile-verified, matrix operand order checked by reading, not run on hardware
//
#include <hip/hip_runtime.h>

// ---------------------------------------------------------------------------
// GCRN forward for MI455X (gfx1250): bf16 WMMA + TDM (tensor_load_to_lds)
// double-buffered LDS staging of the shared B tile, fp32 accumulate.
// ---------------------------------------------------------------------------

#define DEV __device__ __forceinline__

typedef __bf16 bf16;
typedef __bf16 v16bf __attribute__((ext_vector_type(16)));
typedef float  v8f   __attribute__((ext_vector_type(8)));
typedef float  f4v   __attribute__((ext_vector_type(4)));
typedef unsigned int u32x4 __attribute__((ext_vector_type(4)));
typedef int    i32x4 __attribute__((ext_vector_type(4)));
typedef int    i32x8 __attribute__((ext_vector_type(8)));

// Problem constants (from reference)
constexpr int kB   = 16;
constexpr int kN   = 1024;
constexpr int kF   = 256;
constexpr int kG   = 256;
constexpr int kE   = 4;
constexpr int kNE  = 16384;
constexpr int kH1  = 512;
constexpr int kH2  = 512;
constexpr int kEMB = 256;
constexpr int kBN  = kB * kN;     // 16384 MLP rows
constexpr int kDIN = kE * kG;     // 1024
constexpr float kBnEps = 1e-5f;

// ---------------------------------------------------------------------------
// Helpers
// ---------------------------------------------------------------------------
DEV float bfbits_to_f(unsigned short h) {
    unsigned int u = ((unsigned int)h) << 16;
    float f;
    __builtin_memcpy(&f, &u, 4);
    return f;
}
DEV unsigned short f_to_bfbits(float f) {
    bf16 b = (bf16)f;
    unsigned short s;
    __builtin_memcpy(&s, &b, 2);
    return s;
}

// CAS-based bf16 += val
DEV void atomic_add_bf16(bf16* p, float val) {
    size_t ip = (size_t)p;
    unsigned int* base = (unsigned int*)(ip & ~(size_t)3);
    const bool hi = (ip & 2) != 0;
    unsigned int old = *base, assumed;
    do {
        assumed = old;
        unsigned short cur = hi ? (unsigned short)(assumed >> 16)
                                : (unsigned short)(assumed & 0xFFFFu);
        unsigned short nh = f_to_bfbits(bfbits_to_f(cur) + val);
        unsigned int nw = hi ? ((assumed & 0x0000FFFFu) | ((unsigned int)nh << 16))
                             : ((assumed & 0xFFFF0000u) | (unsigned int)nh);
        old = atomicCAS(base, assumed, nw);
    } while (old != assumed);
}

// Low 32 bits of a generic LDS pointer == byte offset within the LDS aperture
DEV unsigned lds_byte_off(const void* p) {
    return (unsigned)(unsigned long long)p;
}

// ---------------------------------------------------------------------------
// TDM: issue a 2-D tile load Global -> LDS via tensor_load_to_lds.
// D# packing per CDNA5 ISA:
//   group0: [1:0]=count=1 | lds_addr @ [63:32] | global_addr @ [120:64] | type=2 @ [127:126]
//   group1: data_size @ [17:16] | tensor_dim0 @ [79:48] | tensor_dim1 @ [111:80]
//           | tile_dim0 @ [127:112] | tile_dim1 @ [143:128] | dim0_stride @ [207:160]
// 2-D tile => groups 2/3 zero (tile_dim2 = 0 -> unused).
// data_size = 1 (2-byte elements). workgroup_mask = 0 (not in a cluster).
// ---------------------------------------------------------------------------
DEV void tdm_load_tile_2d(unsigned lds_off, const void* gaddr,
                          unsigned tile_w, unsigned tile_h,
                          unsigned row_stride_elems)
{
    unsigned long long ga = (unsigned long long)gaddr;
    u32x4 g0;
    g0[0] = 1u;                                        // count=1, user descriptor
    g0[1] = lds_off;                                   // lds_addr (bytes)
    g0[2] = (unsigned)(ga & 0xFFFFFFFFu);              // global_addr[31:0]
    g0[3] = (unsigned)((ga >> 32) & 0x01FFFFFFu)       // global_addr[56:32]
          | (2u << 30);                                // type = 2 ("image")

    i32x8 g1;
    g1[0] = (int)(1u << 16);                           // data_size = 2 bytes
    g1[1] = (int)((tile_w & 0xFFFFu) << 16);           // tensor_dim0[15:0] @ 63:48
    g1[2] = (int)(((tile_w >> 16) & 0xFFFFu)           // tensor_dim0[31:16]
          |       ((tile_h & 0xFFFFu) << 16));         // tensor_dim1[15:0]
    g1[3] = (int)(((tile_h >> 16) & 0xFFFFu)           // tensor_dim1[31:16]
          |       ((tile_w & 0xFFFFu) << 16));         // tile_dim0
    g1[4] = (int)(tile_h & 0xFFFFu);                   // tile_dim1 (tile_dim2 = 0)
    g1[5] = (int)row_stride_elems;                     // tensor_dim0_stride[31:0]
    g1[6] = 0;                                         // stride[47:32], dim1_stride lo
    g1[7] = 0;

    i32x4 gz = {0, 0, 0, 0};
#if defined(__clang_major__) && (__clang_major__ >= 23)
    i32x8 gz8 = {0, 0, 0, 0, 0, 0, 0, 0};
    __builtin_amdgcn_tensor_load_to_lds(g0, g1, gz, gz, gz8, 0);
#else
    __builtin_amdgcn_tensor_load_to_lds(g0, g1, gz, gz, 0);
#endif
}

// ---------------------------------------------------------------------------
// WMMA fragment loads (16-bit A-layout):
//   lanes 0-15 : row = r0+lane,    K = [k0..k0+7] and [k0+16..k0+23]
//   lanes 16-31: row = r0+lane-16, K = [k0+8..k0+15] and [k0+24..k0+31]
// ---------------------------------------------------------------------------
DEV v16bf load_frag(const bf16* __restrict__ base, int ld, int r0, int k0, int lane) {
    const int r  = r0 + (lane & 15);
    const int kk = k0 + ((lane & 16) ? 8 : 0);
    const bf16* p = base + (size_t)r * ld + kk;
    union { v16bf v; f4v q[2]; } u;
    u.q[0] = *reinterpret_cast<const f4v*>(p);
    u.q[1] = *reinterpret_cast<const f4v*>(p + 16);
    return u.v;
}

// B fragment from the LDS-staged 64x32 tile (row-major, 32 elems/row)
DEV v16bf load_frag_lds(const bf16* base, int r0, int lane) {
    const int r  = r0 + (lane & 15);
    const int kk = (lane & 16) ? 8 : 0;
    const bf16* p = base + r * 32 + kk;
    union { v16bf v; f4v q[2]; } u;
    u.q[0] = *reinterpret_cast<const f4v*>(p);
    u.q[1] = *reinterpret_cast<const f4v*>(p + 16);
    return u.v;
}

DEV v8f wmma_bf16(v16bf a, v16bf b, v8f c) {
    return __builtin_amdgcn_wmma_f32_16x16x32_bf16(
        false, a, false, b, (short)0, c, false, false);
}

// ---------------------------------------------------------------------------
// GEMM:  C(MxNc) = A(MxK) @ Bt(NcxK)^T   (bf16 in, fp32 acc)
// Block = 8 waves = 256x64 C tile; each wave owns 32x64 (2x4 WMMA accums).
// The 64x32 B tile is TDM-staged into LDS (double buffered) and shared by
// all 8 waves; A fragments stream from global per wave.
//
// MODE 0: MLP layer.  C fp32 = acc + bias[n]; ldc given. (z == 0)
// MODE 1: XW stage.   z=b*kE+e. A=Xb+b*N*F, Bt=WsT+e*G*F.
//                     Store bf16 TRANSPOSED: XWt[z][n][m] (G x N).
// MODE 2: Aggregation. A=adj+z*N*N, Bt=XWt+z*G*N.
//                     Store bf16 relu into Hb[(b*N+m)*kDIN + e*G + n].
// ---------------------------------------------------------------------------
template <int MODE>
__global__ __launch_bounds__(256) void gemm_wmma_kernel(
    const bf16* __restrict__ Aall, const bf16* __restrict__ Ball,
    void* __restrict__ Call, const float* __restrict__ bias,
    int M, int Nc, int K, int ldc)
{
    __shared__ bf16 sB[2][64 * 32];                  // 2 x 4 KB

    const int lane   = (int)threadIdx.x & 31;
    const int wv     = (int)threadIdx.x >> 5;        // 0..7
    const int tilesN = Nc >> 6;
    const int tm     = blockIdx.x / tilesN;
    const int tn     = blockIdx.x % tilesN;
    const int m0     = tm * 256 + wv * 32;
    const int n0     = tn * 64;
    const int z      = blockIdx.z;

    const bf16* A;
    const bf16* Bt;
    if (MODE == 1) {
        A  = Aall + (size_t)(z >> 2) * kN * kF;      // per-b X
        Bt = Ball + (size_t)(z & 3) * kG * kF;       // per-e WsT (G x F)
    } else if (MODE == 2) {
        A  = Aall + (size_t)z * kN * kN;             // adjacency (N x N)
        Bt = Ball + (size_t)z * kG * kN;             // XWt (G x N)
    } else {
        A  = Aall;
        Bt = Ball;
    }
    const bf16* Btile = Bt + (size_t)n0 * K;         // 64 rows of Bt

    v8f acc[2][4];
#pragma unroll
    for (int i = 0; i < 2; ++i)
#pragma unroll
        for (int j = 0; j < 4; ++j)
#pragma unroll
            for (int t = 0; t < 8; ++t) acc[i][j][t] = 0.0f;

    // Prologue: wave 0 issues TDM for the first B tile
    if (threadIdx.x < 32) {
        tdm_load_tile_2d(lds_byte_off(&sB[0][0]), Btile, 32, 64, (unsigned)K);
    }

    int buf = 0;
    for (int k = 0; k < K; k += 32) {
        if (threadIdx.x < 32) {
            if (k + 32 < K) {
                tdm_load_tile_2d(lds_byte_off(&sB[buf ^ 1][0]),
                                 Btile + (k + 32), 32, 64, (unsigned)K);
                __builtin_amdgcn_s_wait_tensorcnt(1);   // tile k has landed
            } else {
                __builtin_amdgcn_s_wait_tensorcnt(0);
            }
        }
        __syncthreads();                                // publish LDS tile

        if (k + 32 < K) {   // gfx1250 global_prefetch_b8: next A tile
            __builtin_prefetch(A + (size_t)(m0 + (lane & 15)) * K + k + 32);
        }
        v16bf a0 = load_frag(A, K, m0,      k, lane);
        v16bf a1 = load_frag(A, K, m0 + 16, k, lane);
#pragma unroll
        for (int j = 0; j < 4; ++j) {
            v16bf b = load_frag_lds(&sB[buf][0], 16 * j, lane);
            acc[0][j] = wmma_bf16(a0, b, acc[0][j]);
            acc[1][j] = wmma_bf16(a1, b, acc[1][j]);
        }
        __syncthreads();                                // tile consumed
        buf ^= 1;
    }

    // ---- store: element t of lane -> m_local = t + (lane>=16?8:0), n_local = lane&15
    const int col = lane & 15;
    const int rof = (lane & 16) ? 8 : 0;

#pragma unroll
    for (int i = 0; i < 2; ++i) {
#pragma unroll
        for (int j = 0; j < 4; ++j) {
#pragma unroll
            for (int t = 0; t < 8; ++t) {
                const int m = m0 + 16 * i + rof + t;
                const int n = n0 + 16 * j + col;
                const float v = acc[i][j][t];
                if (MODE == 0) {
                    float* C = (float*)Call;
                    C[(size_t)m * ldc + n] = v + bias[n];
                } else if (MODE == 1) {
                    bf16* Ct = (bf16*)Call + (size_t)z * kG * kN;
                    Ct[(size_t)n * kN + m] = (bf16)v;            // transposed
                } else {  // MODE 2: relu -> Hb[(b*N+m)*kDIN + e*G + n]
                    bf16* C = (bf16*)Call
                            + ((size_t)(z >> 2) * kN) * kDIN
                            + (size_t)(z & 3) * kG;
                    C[(size_t)m * kDIN + n] = (bf16)(v > 0.0f ? v : 0.0f);
                }
            }
        }
    }
}

// ---------------------------------------------------------------------------
// Small utility kernels
// ---------------------------------------------------------------------------
__global__ void f32_to_bf16_kernel(const float* __restrict__ s,
                                   bf16* __restrict__ d, size_t n) {
    size_t i = (size_t)blockIdx.x * blockDim.x + threadIdx.x;
    if (i < n) d[i] = (bf16)s[i];
}

// Ws (E,F,G) -> WsT (E,G,F) bf16
__global__ void ws_transpose_kernel(const float* __restrict__ Ws,
                                    bf16* __restrict__ WsT) {
    size_t i = (size_t)blockIdx.x * blockDim.x + threadIdx.x;
    if (i >= (size_t)kE * kG * kF) return;
    int e = (int)(i / (kG * kF));
    int r = (int)(i % (kG * kF));
    int g = r / kF;
    int f = r % kF;
    WsT[i] = (bf16)Ws[(size_t)e * kF * kG + (size_t)f * kG + g];
}

__global__ void zero16_kernel(f4v* __restrict__ p, size_t n16) {
    size_t i = (size_t)blockIdx.x * blockDim.x + threadIdx.x;
    f4v z = {0.0f, 0.0f, 0.0f, 0.0f};
    if (i < n16) p[i] = z;
}

// 1M edges -> bf16 dense adjacency (duplicate edges accumulate exactly)
__global__ void scatter_edges_kernel(const int* __restrict__ A,
                                     bf16* __restrict__ adj) {
    size_t t = (size_t)blockIdx.x * blockDim.x + threadIdx.x;
    if (t >= (size_t)kB * kE * kNE) return;
    int be = (int)(t / kNE);
    int j  = (int)(t % kNE);
    const int* base = A + (size_t)be * 2 * kNE;
    int row = base[j];
    int col = base[kNE + j];
    atomic_add_bf16(adj + ((size_t)be * kN + row) * kN + col, 1.0f);
}

// Column-wise sum / sum-of-squares over kBN rows (fp32, atomic partials)
__global__ __launch_bounds__(256) void bn_stats_kernel(
    const float* __restrict__ h, float* __restrict__ sum,
    float* __restrict__ sumsq, int C)
{
    const int col   = blockIdx.x * 64 + ((int)threadIdx.x & 63);
    const int rlane = (int)threadIdx.x >> 6;                 // 0..3
    const int r0    = blockIdx.y * 256 + rlane;
    float s = 0.0f, s2 = 0.0f;
#pragma unroll 4
    for (int it = 0; it < 64; ++it) {
        float v = h[(size_t)(r0 + it * 4) * C + col];
        s  += v;
        s2 += v * v;
    }
    atomicAdd(&sum[col], s);
    atomicAdd(&sumsq[col], s2);
}

// normalize -> scale/shift -> ELU -> bf16 (feeds next WMMA GEMM)
__global__ void bn_apply_kernel(const float* __restrict__ h,
                                const float* __restrict__ sum,
                                const float* __restrict__ sumsq,
                                const float* __restrict__ g,
                                const float* __restrict__ beta,
                                bf16* __restrict__ out, int C, size_t total)
{
    size_t i = (size_t)blockIdx.x * blockDim.x + threadIdx.x;
    if (i >= total) return;
    int col = (int)(i % C);
    const float inv_n = 1.0f / (float)kBN;
    float m   = sum[col] * inv_n;
    float var = sumsq[col] * inv_n - m * m;
    float y   = (h[i] - m) * rsqrtf(var + kBnEps) * g[col] + beta[col];
    y = (y > 0.0f) ? y : (__expf(y) - 1.0f);
    out[i] = (bf16)y;
}

// ---------------------------------------------------------------------------
// Host-side orchestration
// ---------------------------------------------------------------------------
extern "C" void kernel_launch(void* const* d_in, const int* in_sizes, int n_in,
                              void* d_out, int out_size, void* d_ws, size_t ws_size,
                              hipStream_t stream) {
    (void)in_sizes; (void)n_in; (void)out_size; (void)ws_size;

    const int*   A   = (const int*)  d_in[0];
    const float* X   = (const float*)d_in[1];
    const float* Ws  = (const float*)d_in[2];
    const float* W0  = (const float*)d_in[3];
    const float* b0  = (const float*)d_in[4];
    const float* g0  = (const float*)d_in[5];
    const float* be0 = (const float*)d_in[6];
    const float* W1  = (const float*)d_in[7];
    const float* b1  = (const float*)d_in[8];
    const float* g1  = (const float*)d_in[9];
    const float* be1 = (const float*)d_in[10];
    const float* W2  = (const float*)d_in[11];
    const float* b2  = (const float*)d_in[12];
    float* out = (float*)d_out;

    // ---- workspace arena (256B aligned slices) ----
    char* w = (char*)d_ws;
    size_t off = 0;
    auto alloc = [&](size_t bytes) -> void* {
        void* p = w + off;
        off = (off + bytes + 255) & ~(size_t)255;
        return p;
    };
    bf16*  Xb    = (bf16*) alloc((size_t)kB * kN * kF * 2);        //   8 MB
    bf16*  WsT   = (bf16*) alloc((size_t)kE * kG * kF * 2);        // 0.5 MB
    bf16*  W0b   = (bf16*) alloc((size_t)kH1 * kDIN * 2);          //   1 MB
    bf16*  W1b   = (bf16*) alloc((size_t)kH2 * kH1 * 2);           // 0.5 MB
    bf16*  W2b   = (bf16*) alloc((size_t)kEMB * kH2 * 2);          // .25 MB
    bf16*  XWt   = (bf16*) alloc((size_t)kB * kE * kG * kN * 2);   //  32 MB
    bf16*  Hb    = (bf16*) alloc((size_t)kBN * kDIN * 2);          //  32 MB
    float* stats = (float*)alloc((size_t)4 * kH1 * 4);             //  8 KB
    bf16*  adjB  = (bf16*) alloc((size_t)kB * kE * kN * kN * 2);   // 128 MB
    // MLP intermediates overlay the (dead-after-aggregation) adjacency:
    float* h1f = (float*)adjB;                                     //  32 MB
    bf16*  h1b = (bf16*) ((char*)adjB + (size_t)32 * 1024 * 1024); //  16 MB
    float* h2f = (float*)((char*)adjB + (size_t)48 * 1024 * 1024); //  32 MB
    bf16*  h2b = (bf16*) ((char*)adjB + (size_t)80 * 1024 * 1024); //  16 MB
    float* sum0 = stats, *sq0 = stats + kH1, *sum1 = stats + 2 * kH1, *sq1 = stats + 3 * kH1;

    const int T = 256;

    // 1) fp32 -> bf16 conversions (X, weights) + Ws transpose
    {
        size_t n = (size_t)kB * kN * kF;
        f32_to_bf16_kernel<<<dim3((unsigned)((n + T - 1) / T)), T, 0, stream>>>(X, Xb, n);
    }
    ws_transpose_kernel<<<dim3((kE * kG * kF + T - 1) / T), T, 0, stream>>>(Ws, WsT);
    {
        size_t n = (size_t)kH1 * kDIN;
        f32_to_bf16_kernel<<<dim3((unsigned)((n + T - 1) / T)), T, 0, stream>>>(W0, W0b, n);
        n = (size_t)kH2 * kH1;
        f32_to_bf16_kernel<<<dim3((unsigned)((n + T - 1) / T)), T, 0, stream>>>(W1, W1b, n);
        n = (size_t)kEMB * kH2;
        f32_to_bf16_kernel<<<dim3((unsigned)((n + T - 1) / T)), T, 0, stream>>>(W2, W2b, n);
    }

    // 2) zero adjacency + stats, scatter edges
    {
        size_t n16 = ((size_t)kB * kE * kN * kN * 2) / 16;
        zero16_kernel<<<dim3((unsigned)((n16 + T - 1) / T)), T, 0, stream>>>((f4v*)adjB, n16);
        size_t s16 = ((size_t)4 * kH1 * 4) / 16;
        zero16_kernel<<<dim3((unsigned)((s16 + T - 1) / T)), T, 0, stream>>>((f4v*)stats, s16);
    }
    {
        size_t n = (size_t)kB * kE * kNE;
        scatter_edges_kernel<<<dim3((unsigned)((n + T - 1) / T)), T, 0, stream>>>(A, adjB);
    }

    // 3) XW = X @ Ws  (per b,e) -> XWt (transposed bf16)
    //    M=1024 (4 Mtiles), Nc=256 (4 Ntiles) -> 16 blocks, z=64
    gemm_wmma_kernel<1><<<dim3(16, 1, kB * kE), T, 0, stream>>>(
        Xb, WsT, XWt, nullptr, kN, kG, kF, 0);

    // 4) H = relu(adj @ XW) -> Hb (bf16, (B*N) x 1024); K=1024
    gemm_wmma_kernel<2><<<dim3(16, 1, kB * kE), T, 0, stream>>>(
        adjB, XWt, Hb, nullptr, kN, kG, kN, 0);

    // 5) L0: h1 = Hb @ W0^T + b0   (16384x1024 @ 1024x512): 64*8 = 512 blocks
    gemm_wmma_kernel<0><<<dim3(512, 1, 1), T, 0, stream>>>(
        Hb, W0b, h1f, b0, kBN, kH1, kDIN, kH1);

    // 6) BN + ELU -> h1b (bf16)
    bn_stats_kernel<<<dim3(kH1 / 64, kBN / 256), T, 0, stream>>>(h1f, sum0, sq0, kH1);
    {
        size_t n = (size_t)kBN * kH1;
        bn_apply_kernel<<<dim3((unsigned)((n + T - 1) / T)), T, 0, stream>>>(
            h1f, sum0, sq0, g0, be0, h1b, kH1, n);
    }

    // 7) L1: h2 = h1b @ W1^T + b1   (16384x512 @ 512x512): 512 blocks
    gemm_wmma_kernel<0><<<dim3(512, 1, 1), T, 0, stream>>>(
        h1b, W1b, h2f, b1, kBN, kH2, kH1, kH2);

    // 8) BN + ELU -> h2b
    bn_stats_kernel<<<dim3(kH2 / 64, kBN / 256), T, 0, stream>>>(h2f, sum1, sq1, kH2);
    {
        size_t n = (size_t)kBN * kH2;
        bn_apply_kernel<<<dim3((unsigned)((n + T - 1) / T)), T, 0, stream>>>(
            h2f, sum1, sq1, g1, be1, h2b, kH2, n);
    }

    // 9) L2: out = h2b @ W2^T + b2  (16384x512 @ 512x256): 256 blocks
    gemm_wmma_kernel<0><<<dim3(256, 1, 1), T, 0, stream>>>(
        h2b, W2b, out, b2, kBN, kEMB, kH2, kEMB);
}
